// ChildSumLSTMLayer2_57045755625483
// MI455X (gfx1250) — matrix operand order repaired and checked by reading, not compiled, not run on hardware
//
#include <hip/hip_runtime.h>
#include <hip/hip_bf16.h>
#include <math.h>

// ---------------------------------------------------------------------------
// Types for CDNA5 (gfx1250) WMMA: wave32, 16x16x32 bf16 -> f32 accumulate.
// ---------------------------------------------------------------------------
typedef __bf16 bf16_t;
typedef __attribute__((ext_vector_type(16))) __bf16 v16bf;
typedef __attribute__((ext_vector_type(8)))  __bf16 v8bf;
typedef __attribute__((ext_vector_type(8)))  float  v8f;

struct __align__(16) F4 { float  v[4]; };
struct __align__(8)  H4 { bf16_t v[4]; };

#define DHID 128   // DIN == DOUT == 128 in this problem

__device__ __forceinline__ float sigmoidf_(float x) {
  return 1.0f / (1.0f + __expf(-x));
}
__device__ __forceinline__ float tanhf_(float x) {
  float e = __expf(-2.0f * fabsf(x));
  float t = (1.0f - e) / (1.0f + e);
  return copysignf(t, x);
}

// A-fragment (16x32 bf16, M x K) from a 16x128 row-major bf16 LDS tile.
// ISA 7.12.2 16-bit A layout: lanes 0-15 -> M=lane, K=kstep*32 + {0..7,16..23}
//                             lanes 16-31 -> M=lane-16, K=kstep*32 + {8..15,24..31}
__device__ __forceinline__ v16bf make_afrag(const bf16_t* tile, int lane, int kstep) {
  int m  = lane & 15;
  int hi = lane >> 4;
  const bf16_t* row = tile + m * 128 + kstep * 32 + hi * 8;
  union { v16bf v; v8bf h[2]; } u;
  u.h[0] = *(const v8bf*)(row);        // K = base + 0..7
  u.h[1] = *(const v8bf*)(row + 16);   // K = base + 16..23
  return u.v;
}

// B-fragment from pre-swizzled weight buffer: lane's 16 bf16 are contiguous.
__device__ __forceinline__ v16bf load_bfrag(const bf16_t* frags, int ct, int ks, int lane) {
  return *(const v16bf*)(frags + ((size_t)((ct * 4 + ks) * 32 + lane)) * 16);
}

#define WMMA_BF16(A, B, C) \
  __builtin_amdgcn_wmma_f32_16x16x32_bf16(false, (A), false, (B), (short)0, (C), false, false)

// ---------------------------------------------------------------------------
// k_zero: clear accumulators (h_sum, branch_f, flag)
// ---------------------------------------------------------------------------
__global__ void k_zero(float* __restrict__ h_sum, float* __restrict__ branch_f,
                       int* __restrict__ flag, int N) {
  int total = N * DHID;
  for (int i = blockIdx.x * blockDim.x + threadIdx.x; i < total;
       i += gridDim.x * blockDim.x) {
    h_sum[i]    = 0.0f;
    branch_f[i] = 0.0f;
    if (i < N) flag[i] = 0;
  }
}

// ---------------------------------------------------------------------------
// k_swizzle: f32 weight (128 x ncols, row-major) -> bf16 WMMA B-fragment layout.
// dst index = (((ct*4+ks)*32 + lane)*16 + j); value = src[K][N] with
//   N = ct*16 + (lane&15),  K = ks*32 + (lane>>4)*8 + (j&7) + (j>>3)*16   (ISA B layout)
// ---------------------------------------------------------------------------
__global__ void k_swizzle(const float* __restrict__ src, bf16_t* __restrict__ dst,
                          int ncols) {
  int idx = blockIdx.x * blockDim.x + threadIdx.x;
  int total = DHID * ncols;
  if (idx >= total) return;
  int j    = idx & 15;
  int lane = (idx >> 4) & 31;
  int fs   = idx >> 9;          // ct*4 + ks
  int ks   = fs & 3;
  int ct   = fs >> 2;
  int n = ct * 16 + (lane & 15);
  int k = ks * 32 + (lane >> 4) * 8 + (j & 7) + (j >> 3) * 16;
  dst[idx] = (bf16_t)src[(size_t)k * ncols + n];
}

// ---------------------------------------------------------------------------
// k_gemm_wx: Wx = x @ W   (N x 128) @ (128 x 512) -> (N x 512) f32
// One block (128 thr = 4 waves) per 16-node tile; each wave does 8 col-tiles.
// ---------------------------------------------------------------------------
__global__ void k_gemm_wx(const float* __restrict__ x, const bf16_t* __restrict__ wfrag,
                          float* __restrict__ Wx, int N) {
  __shared__ bf16_t lds_x[16 * 128];
  int tid   = threadIdx.x;
  int lane  = tid & 31;
  int wslot = tid >> 5;
  int n0 = blockIdx.x * 16;
  bool full = (n0 + 16 <= N);

  // stage 16x128 f32 -> bf16 LDS, vectorized (128 thr * F4 * 4 iters = 2048)
  if (full) {
#pragma unroll
    for (int it = 0; it < 4; ++it) {
      int flat = it * 512 + tid * 4;
      int r = flat >> 7, c = flat & 127;
      F4 v = *(const F4*)(x + (size_t)(n0 + r) * 128 + c);
      H4 h;
      h.v[0] = (bf16_t)v.v[0]; h.v[1] = (bf16_t)v.v[1];
      h.v[2] = (bf16_t)v.v[2]; h.v[3] = (bf16_t)v.v[3];
      *(H4*)(lds_x + flat) = h;
    }
  } else {
#pragma unroll
    for (int it = 0; it < 4; ++it) {
      int flat = it * 512 + tid * 4;
      int r = flat >> 7, c = flat & 127;
      F4 v = {{0.0f, 0.0f, 0.0f, 0.0f}};
      if (n0 + r < N) v = *(const F4*)(x + (size_t)(n0 + r) * 128 + c);
      H4 h;
      h.v[0] = (bf16_t)v.v[0]; h.v[1] = (bf16_t)v.v[1];
      h.v[2] = (bf16_t)v.v[2]; h.v[3] = (bf16_t)v.v[3];
      *(H4*)(lds_x + flat) = h;
    }
  }
  __syncthreads();

  v16bf a0 = make_afrag(lds_x, lane, 0);
  v16bf a1 = make_afrag(lds_x, lane, 1);
  v16bf a2 = make_afrag(lds_x, lane, 2);
  v16bf a3 = make_afrag(lds_x, lane, 3);

  int m_lo = lane & 15;
  int hi   = lane >> 4;
  for (int ct8 = 0; ct8 < 8; ++ct8) {
    int ct = wslot * 8 + ct8;
    // issue all B-fragment loads before the WMMA chain
    v16bf b0 = load_bfrag(wfrag, ct, 0, lane);
    v16bf b1 = load_bfrag(wfrag, ct, 1, lane);
    v16bf b2 = load_bfrag(wfrag, ct, 2, lane);
    v16bf b3 = load_bfrag(wfrag, ct, 3, lane);
    v8f acc = {};
    acc = WMMA_BF16(a0, b0, acc);
    acc = WMMA_BF16(a1, b1, acc);
    acc = WMMA_BF16(a2, b2, acc);
    acc = WMMA_BF16(a3, b3, acc);
    int col = ct * 16 + m_lo;
    if (full) {
#pragma unroll
      for (int j = 0; j < 8; ++j)
        Wx[(size_t)(n0 + j + 8 * hi) * 512 + col] = acc[j];
    } else {
#pragma unroll
      for (int j = 0; j < 8; ++j) {
        int node = n0 + j + 8 * hi;
        if (node < N) Wx[(size_t)node * 512 + col] = acc[j];
      }
    }
  }
}

// ---------------------------------------------------------------------------
// k_edge: per wave: 128 contiguous edges (seg sorted).
//   T = child_h_tile @ U_f (WMMA) -> LDS
//   column pass: f = sigmoid(T + Wf[seg]); segmented sums of f*child_c and
//   child_h in registers; interior segments -> plain stores, boundary -> atomics
// ---------------------------------------------------------------------------
#define EDGE_TILES_PER_WAVE 8

__global__ void k_edge(const float* __restrict__ child_h, const float* __restrict__ child_c,
                       const int* __restrict__ seg, const float* __restrict__ Wx,
                       const bf16_t* __restrict__ uffrag,
                       float* __restrict__ h_sum, float* __restrict__ branch_f,
                       int* __restrict__ flag, int E) {
  __shared__ bf16_t lds_a[4][16 * 128];   // child_h tile (bf16)
  __shared__ float  lds_t[4][16 * 128];   // GEMM result tile (f32)
  __shared__ int    lds_seg[4][16];

  int tid   = threadIdx.x;
  int lane  = tid & 31;
  int wslot = tid >> 5;
  long wgid   = (long)blockIdx.x * 4 + wslot;
  long e_base = wgid * (EDGE_TILES_PER_WAVE * 16);
  if (e_base >= E) return;

  bf16_t* la  = lds_a[wslot];
  float*  lt  = lds_t[wslot];
  int*    lsg = lds_seg[wslot];
  int c0 = lane * 4;
  int m_lo = lane & 15;
  int hi   = lane >> 4;

  int   first_seg = seg[e_base];
  int   cur_seg   = -1;
  float accH[4] = {0, 0, 0, 0};
  float accF[4] = {0, 0, 0, 0};

  for (int t = 0; t < EDGE_TILES_PER_WAVE; ++t) {
    long e0 = e_base + (long)t * 16;
    if (e0 >= E) break;
    bool full = (e0 + 16 <= E);

    // ---- stage child_h tile as bf16, and seg values
    if (full) {
#pragma unroll
      for (int r = 0; r < 16; ++r) {
        F4 v = *(const F4*)(child_h + (size_t)(e0 + r) * 128 + c0);
        H4 h;
        h.v[0] = (bf16_t)v.v[0]; h.v[1] = (bf16_t)v.v[1];
        h.v[2] = (bf16_t)v.v[2]; h.v[3] = (bf16_t)v.v[3];
        *(H4*)(la + r * 128 + c0) = h;
      }
      if (lane < 16) lsg[lane] = seg[e0 + lane];
    } else {
#pragma unroll
      for (int r = 0; r < 16; ++r) {
        long e = e0 + r;
        F4 v = {{0.0f, 0.0f, 0.0f, 0.0f}};
        if (e < E) v = *(const F4*)(child_h + (size_t)e * 128 + c0);
        H4 h;
        h.v[0] = (bf16_t)v.v[0]; h.v[1] = (bf16_t)v.v[1];
        h.v[2] = (bf16_t)v.v[2]; h.v[3] = (bf16_t)v.v[3];
        *(H4*)(la + r * 128 + c0) = h;
      }
      if (lane < 16) lsg[lane] = (e0 + lane < E) ? seg[e0 + lane] : -1;
    }
    __builtin_amdgcn_wave_barrier();

    // ---- prefetch next tile's streams (global_prefetch_b8); 32 lanes * 256B
    {
      long pn = e0 + 16;
      if (pn + 16 <= E) {
        __builtin_prefetch((const char*)(child_h + pn * 128) + lane * 256, 0, 1);
        __builtin_prefetch((const char*)(child_c + pn * 128) + lane * 256, 0, 1);
      }
    }

    // ---- GEMM: 16x128 = A(16x128) @ U_f(128x128)
    v16bf a0 = make_afrag(la, lane, 0);
    v16bf a1 = make_afrag(la, lane, 1);
    v16bf a2 = make_afrag(la, lane, 2);
    v16bf a3 = make_afrag(la, lane, 3);
#pragma unroll
    for (int ct = 0; ct < 8; ++ct) {
      v16bf b0 = load_bfrag(uffrag, ct, 0, lane);
      v16bf b1 = load_bfrag(uffrag, ct, 1, lane);
      v16bf b2 = load_bfrag(uffrag, ct, 2, lane);
      v16bf b3 = load_bfrag(uffrag, ct, 3, lane);
      v8f acc = {};
      acc = WMMA_BF16(a0, b0, acc);
      acc = WMMA_BF16(a1, b1, acc);
      acc = WMMA_BF16(a2, b2, acc);
      acc = WMMA_BF16(a3, b3, acc);
#pragma unroll
      for (int j = 0; j < 8; ++j)
        lt[(j + 8 * hi) * 128 + ct * 16 + m_lo] = acc[j];
    }
    __builtin_amdgcn_wave_barrier();

    // ---- column pass: lane owns cols [c0, c0+4)
    for (int r = 0; r < 16; ++r) {
      int s = lsg[r];                 // uniform broadcast
      if (s < 0) continue;
      if (s != cur_seg) {
        if (cur_seg >= 0) {
          bool at = (cur_seg == first_seg);   // may be shared with prev wave
#pragma unroll
          for (int i = 0; i < 4; ++i) {
            size_t o = (size_t)cur_seg * 128 + c0 + i;
            if (at) { atomicAdd(&h_sum[o], accH[i]); atomicAdd(&branch_f[o], accF[i]); }
            else    { h_sum[o] = accH[i];            branch_f[o] = accF[i]; }
          }
        }
        cur_seg = s;
        accH[0] = accH[1] = accH[2] = accH[3] = 0.0f;
        accF[0] = accF[1] = accF[2] = accF[3] = 0.0f;
        if (lane == 0) flag[s] = 1;
      }
      long e = e0 + r;
      F4 tv = *(const F4*)(lt + r * 128 + c0);
      F4 wf = *(const F4*)(Wx + (size_t)s * 512 + c0);          // W_f columns
      F4 cc = *(const F4*)(child_c + (size_t)e * 128 + c0);
      H4 ch = *(const H4*)(la + r * 128 + c0);
#pragma unroll
      for (int i = 0; i < 4; ++i) {
        float f = sigmoidf_(tv.v[i] + wf.v[i]);
        accF[i] += f * cc.v[i];
        accH[i] += (float)ch.v[i];
      }
    }
    __builtin_amdgcn_wave_barrier();   // lds_a / lds_t reused next tile
  }

  // final flush: segment may continue into the next wave's range -> atomic
  if (cur_seg >= 0) {
#pragma unroll
    for (int i = 0; i < 4; ++i) {
      size_t o = (size_t)cur_seg * 128 + c0 + i;
      atomicAdd(&h_sum[o], accH[i]);
      atomicAdd(&branch_f[o], accF[i]);
    }
  }
}

// ---------------------------------------------------------------------------
// k_node: iuo = h_sum' @ U_iuo (WMMA) then gate epilogue -> new_c, new_h
//   h_sum' = (flag ? h_sum : h_init)
// Block = 64 threads (2 waves); one 16-node tile per wave.
// ---------------------------------------------------------------------------
__global__ void k_node(const float* __restrict__ Wx, const float* __restrict__ h_sum,
                       const float* __restrict__ branch_f, const int* __restrict__ flag,
                       const float* __restrict__ h_init, const bf16_t* __restrict__ uifrag,
                       float* __restrict__ out_c, float* __restrict__ out_h, int N) {
  __shared__ bf16_t lds_a[2][16 * 128];   // 8 KB
  __shared__ float  lds_t[2][16 * 384];   // 48 KB

  int tid   = threadIdx.x;
  int lane  = tid & 31;
  int wslot = tid >> 5;
  int n0 = (blockIdx.x * 2 + wslot) * 16;
  if (n0 >= N) return;
  bool full = (n0 + 16 <= N);

  bf16_t* la = lds_a[wslot];
  float*  lt = lds_t[wslot];
  int c0 = lane * 4;
  int m_lo = lane & 15;
  int hi   = lane >> 4;

#pragma unroll
  for (int r = 0; r < 16; ++r) {
    int node = n0 + r;
    F4 v = {{0.0f, 0.0f, 0.0f, 0.0f}};
    if (full || node < N) {
      if (flag[node]) v = *(const F4*)(h_sum + (size_t)node * 128 + c0);
      else            v = *(const F4*)(h_init + c0);
    }
    H4 h;
    h.v[0] = (bf16_t)v.v[0]; h.v[1] = (bf16_t)v.v[1];
    h.v[2] = (bf16_t)v.v[2]; h.v[3] = (bf16_t)v.v[3];
    *(H4*)(la + r * 128 + c0) = h;
  }
  __builtin_amdgcn_wave_barrier();

  v16bf a0 = make_afrag(la, lane, 0);
  v16bf a1 = make_afrag(la, lane, 1);
  v16bf a2 = make_afrag(la, lane, 2);
  v16bf a3 = make_afrag(la, lane, 3);

  for (int ct = 0; ct < 24; ++ct) {
    v16bf b0 = load_bfrag(uifrag, ct, 0, lane);
    v16bf b1 = load_bfrag(uifrag, ct, 1, lane);
    v16bf b2 = load_bfrag(uifrag, ct, 2, lane);
    v16bf b3 = load_bfrag(uifrag, ct, 3, lane);
    v8f acc = {};
    acc = WMMA_BF16(a0, b0, acc);
    acc = WMMA_BF16(a1, b1, acc);
    acc = WMMA_BF16(a2, b2, acc);
    acc = WMMA_BF16(a3, b3, acc);
#pragma unroll
    for (int j = 0; j < 8; ++j)
      lt[(j + 8 * hi) * 384 + ct * 16 + m_lo] = acc[j];
  }
  __builtin_amdgcn_wave_barrier();

  for (int r = 0; r < 16; ++r) {
    int node = n0 + r;
    if (!full && node >= N) break;
    F4 vi = *(const F4*)(lt + r * 384 + c0);
    F4 vu = *(const F4*)(lt + r * 384 + 128 + c0);
    F4 vo = *(const F4*)(lt + r * 384 + 256 + c0);
    F4 wi = *(const F4*)(Wx + (size_t)node * 512 + 128 + c0);
    F4 wu = *(const F4*)(Wx + (size_t)node * 512 + 256 + c0);
    F4 wo = *(const F4*)(Wx + (size_t)node * 512 + 384 + c0);
    F4 bf = *(const F4*)(branch_f + (size_t)node * 128 + c0);
    F4 oc, oh;
#pragma unroll
    for (int i = 0; i < 4; ++i) {
      float gi = sigmoidf_(vi.v[i] + wi.v[i]);
      float gu = tanhf_(vu.v[i] + wu.v[i]);
      float go = sigmoidf_(vo.v[i] + wo.v[i]);
      float nc = gi * gu + bf.v[i];
      oc.v[i] = nc;
      oh.v[i] = go * tanhf_(nc);
    }
    *(F4*)(out_c + (size_t)node * 128 + c0) = oc;
    *(F4*)(out_h + (size_t)node * 128 + c0) = oh;
  }
}

// ---------------------------------------------------------------------------
// launcher
// ---------------------------------------------------------------------------
extern "C" void kernel_launch(void* const* d_in, const int* in_sizes, int n_in,
                              void* d_out, int out_size, void* d_ws, size_t ws_size,
                              hipStream_t stream) {
  const float* x       = (const float*)d_in[0];
  const float* child_h = (const float*)d_in[1];
  const float* child_c = (const float*)d_in[2];
  const int*   seg     = (const int*)  d_in[3];
  const float* W       = (const float*)d_in[4];
  const float* U_f     = (const float*)d_in[5];
  const float* U_iuo   = (const float*)d_in[6];
  const float* h_init  = (const float*)d_in[7];

  int N = in_sizes[0] / DHID;   // 50000
  int E = in_sizes[1] / DHID;   // 800000

  char* ws = (char*)d_ws;
  size_t off = 0;
  auto take = [&](size_t bytes) -> char* {
    char* p = ws + off;
    off = (off + bytes + 255) & ~(size_t)255;
    return p;
  };
  float*  Wx       = (float*) take((size_t)N * 512 * 4);
  float*  h_sum    = (float*) take((size_t)N * 128 * 4);
  float*  branch_f = (float*) take((size_t)N * 128 * 4);
  int*    flag     = (int*)   take((size_t)N * 4);
  bf16_t* wfrag    = (bf16_t*)take((size_t)128 * 512 * 2);
  bf16_t* uffrag   = (bf16_t*)take((size_t)128 * 128 * 2);
  bf16_t* uifrag   = (bf16_t*)take((size_t)128 * 384 * 2);

  // 1) zero accumulators
  {
    int total = N * DHID;
    int zb = (total + 255) / 256;
    if (zb > 8192) zb = 8192;
    k_zero<<<zb, 256, 0, stream>>>(h_sum, branch_f, flag, N);
  }
  // 2) weight swizzles (f32 -> bf16 fragment layout)
  k_swizzle<<<(128 * 512 + 255) / 256, 256, 0, stream>>>(W,     wfrag,  512);
  k_swizzle<<<(128 * 128 + 255) / 256, 256, 0, stream>>>(U_f,   uffrag, 128);
  k_swizzle<<<(128 * 384 + 255) / 256, 256, 0, stream>>>(U_iuo, uifrag, 384);

  // 3) Wx = x @ W
  int ntiles = (N + 15) / 16;
  k_gemm_wx<<<ntiles, 128, 0, stream>>>(x, wfrag, Wx, N);

  // 4) edge phase: f_k GEMM + segmented sums
  long waves  = ((long)E + EDGE_TILES_PER_WAVE * 16 - 1) / (EDGE_TILES_PER_WAVE * 16);
  int  eblocks = (int)((waves + 3) / 4);
  k_edge<<<eblocks, 128, 0, stream>>>(child_h, child_c, seg, Wx, uffrag,
                                      h_sum, branch_f, flag, E);

  // 5) node phase: iuo GEMM + gates -> outputs (new_c, then new_h)
  float* out_c = (float*)d_out;
  float* out_h = out_c + (size_t)N * 128;
  int nblocks = (ntiles + 1) / 2;
  k_node<<<nblocks, 64, 0, stream>>>(Wx, h_sum, branch_f, flag, h_init, uifrag,
                                     out_c, out_h, N);
}
